// Int4Linear_28690381537821
// MI455X (gfx1250) — compile-verified, matrix operand order checked by reading, not run on hardware
//
#include <hip/hip_runtime.h>

typedef __attribute__((ext_vector_type(16))) _Float16 v16h;
typedef __attribute__((ext_vector_type(8)))  _Float16 v8h;
typedef __attribute__((ext_vector_type(4)))  _Float16 v4h;
typedef __attribute__((ext_vector_type(8)))  float    v8f;
typedef __attribute__((ext_vector_type(4)))  int      v4i;

#define OUT_F     11008
#define IN_F      4096
#define GROUP_SZ  128
#define NGROUPS   32
#define PACKED_W  2048
#define BATCH_M   32

#define LDS_ROW     136   // 128 + 8 halfs pad: 272B row stride (16B aligned, bank-staggered)
#define WG_THREADS  128   // 4 wave32s
#define COLS_PER_WG 64    // 4 waves * 16 columns -> 172 workgroups

#define XF16_BYTES  (BATCH_M * IN_F * 2)   // 262144

// Pre-pass: convert x (32x4096 f32) to f16 once into workspace.
__global__ __launch_bounds__(256)
void xcast_f16_kernel(const float* __restrict__ x, _Float16* __restrict__ xf)
{
    const int t = blockIdx.x * 256 + threadIdx.x;       // 0..16383
    const float4* __restrict__ xv = (const float4*)x;
    const float4 a = xv[(size_t)t * 2];
    const float4 b = xv[(size_t)t * 2 + 1];
    v8h h;
    h[0] = (_Float16)a.x; h[1] = (_Float16)a.y;
    h[2] = (_Float16)a.z; h[3] = (_Float16)a.w;
    h[4] = (_Float16)b.x; h[5] = (_Float16)b.y;
    h[6] = (_Float16)b.z; h[7] = (_Float16)b.w;
    *(v8h*)(xf + (size_t)t * 8) = h;
}

template<bool XF16>
__global__ __launch_bounds__(WG_THREADS)
void int4linear_wmma_kernel(const float*    __restrict__ x,
                            const _Float16* __restrict__ xf16,
                            const int*      __restrict__ wpack,
                            const float*    __restrict__ scale,
                            const float*    __restrict__ zpoint,
                            const float*    __restrict__ bias,
                            float*          __restrict__ out)
{
    __shared__ _Float16 xs[BATCH_M * LDS_ROW];   // 8704 bytes

    const int tid  = threadIdx.x;
    const int wave = tid >> 5;        // 0..3
    const int lane = tid & 31;
    const int ncol = lane & 15;       // output column within wave tile / B-matrix N
    const int kh   = lane >> 4;       // K-half select

    const int ocol = blockIdx.x * COLS_PER_WG + wave * 16 + ncol;
    const int* __restrict__ wrow = wpack + (size_t)ocol * PACKED_W;
    const float bb = bias[ocol];

    v8f acc0 = {};   // M = 0..15
    v8f acc1 = {};   // M = 16..31

    for (int g = 0; g < NGROUPS; ++g) {
        __syncthreads();   // previous group's LDS reads complete before restage

        // Stage x[:, g*128 : (g+1)*128] into LDS as f16.
        if constexpr (XF16) {
            // Pure b128 copy from pre-converted f16 x in workspace.
            const v8h* __restrict__ xg = (const v8h*)xf16 + g * (GROUP_SZ / 8);
            #pragma unroll
            for (int i = 0; i < 4; ++i) {
                const int p   = tid + i * WG_THREADS;  // 0..511 16B chunks
                const int row = p >> 4;                // 16 chunks per row-group
                const int cp  = p & 15;
                const v8h h = xg[(size_t)row * (IN_F / 8) + cp];
                *(v8h*)&xs[row * LDS_ROW + cp * 8] = h;
            }
        } else {
            // Fallback: convert from f32 on the fly.
            const float4* __restrict__ xg =
                (const float4*)x + (size_t)g * (GROUP_SZ / 4);
            #pragma unroll
            for (int i = 0; i < 8; ++i) {
                const int p   = tid + i * WG_THREADS;  // 0..1023 float4 slots
                const int row = p >> 5;                // 32 float4 per row-group
                const int cp  = p & 31;
                const float4 v = xg[(size_t)row * (IN_F / 4) + cp];
                v4h h;
                h[0] = (_Float16)v.x; h[1] = (_Float16)v.y;
                h[2] = (_Float16)v.z; h[3] = (_Float16)v.w;
                *(v4h*)&xs[row * LDS_ROW + cp * 4] = h;
            }
        }
        __syncthreads();

        const float s  = scale[ocol * NGROUPS + g];
        const float z  = zpoint[ocol * NGROUPS + g];
        const _Float16 hs  = (_Float16)s;
        const _Float16 hms = (_Float16)(-z * s);   // dequant: w = u*s + (-z*s)

        // Prefetch next group's packed weights for this lane's column.
        if (g + 1 < NGROUPS)
            __builtin_prefetch(wrow + (g + 1) * (GROUP_SZ / 2), 0, 1);

        #pragma unroll
        for (int it = 0; it < 4; ++it) {
            const int klocal = it * 32 + kh * 8;       // K offset inside group
            const int kglob  = g * GROUP_SZ + klocal;  // global K

            // Weights are streamed exactly once -> non-temporal b128 loads.
            const v4i w0 = __builtin_nontemporal_load(
                (const v4i*)(wrow + (kglob >> 1)));
            const v4i w1 = __builtin_nontemporal_load(
                (const v4i*)(wrow + ((kglob + 16) >> 1)));

            // B fragment: high nibble = even K, low nibble = odd K.
            // All-f16 dequant: cvt_f16_u16 + pk_fma_f16.
            v16h bf;
            #pragma unroll
            for (int j = 0; j < 4; ++j) {
                const int y0 = w0[j];                  // value in [0,255]
                bf[2 * j]     = (_Float16)(unsigned short)(y0 >> 4) * hs + hms;
                bf[2 * j + 1] = (_Float16)(unsigned short)(y0 & 15) * hs + hms;
                const int y1 = w1[j];
                bf[8 + 2 * j]     = (_Float16)(unsigned short)(y1 >> 4) * hs + hms;
                bf[8 + 2 * j + 1] = (_Float16)(unsigned short)(y1 & 15) * hs + hms;
            }

            // A fragments from LDS: per lane two 16B chunks (K kb..kb+7, kb+16..kb+23).
            union AF { v16h v; v8h h[2]; };
            AF a0, a1;
            const _Float16* xr0 = &xs[ncol * LDS_ROW + klocal];
            a0.h[0] = *(const v8h*)(xr0);
            a0.h[1] = *(const v8h*)(xr0 + 16);
            const _Float16* xr1 = &xs[(ncol + 16) * LDS_ROW + klocal];
            a1.h[0] = *(const v8h*)(xr1);
            a1.h[1] = *(const v8h*)(xr1 + 16);

            acc0 = __builtin_amdgcn_wmma_f32_16x16x32_f16(
                false, a0.v, false, bf, (short)0, acc0, false, false);
            acc1 = __builtin_amdgcn_wmma_f32_16x16x32_f16(
                false, a1.v, false, bf, (short)0, acc1, false, false);
        }
    }

    // Epilogue: C/D layout -> VGPR r holds M = r + 8*kh, N = ncol.
    // Output is never re-read by this kernel -> non-temporal stores.
    #pragma unroll
    for (int r = 0; r < 8; ++r) {
        const int m = r + 8 * kh;
        __builtin_nontemporal_store(acc0[r] + bb, &out[(size_t)m * OUT_F + ocol]);
        __builtin_nontemporal_store(acc1[r] + bb, &out[(size_t)(m + 16) * OUT_F + ocol]);
    }
}

extern "C" void kernel_launch(void* const* d_in, const int* in_sizes, int n_in,
                              void* d_out, int out_size, void* d_ws, size_t ws_size,
                              hipStream_t stream) {
    const float* x     = (const float*)d_in[0];
    const int*   wpack = (const int*)  d_in[1];
    const float* scale = (const float*)d_in[2];
    const float* zp    = (const float*)d_in[3];
    const float* bias  = (const float*)d_in[4];
    float* out = (float*)d_out;

    dim3 grid(OUT_F / COLS_PER_WG);   // 172 workgroups (11008 = 172 * 64 exactly)
    dim3 block(WG_THREADS);           // 4 wave32s, 8.7KB LDS -> multiple WGs per WGP

    if (ws_size >= (size_t)XF16_BYTES) {
        _Float16* xf = (_Float16*)d_ws;
        xcast_f16_kernel<<<dim3(64), dim3(256), 0, stream>>>(x, xf);
        int4linear_wmma_kernel<true><<<grid, block, 0, stream>>>(
            x, xf, wpack, scale, zp, bias, out);
    } else {
        int4linear_wmma_kernel<false><<<grid, block, 0, stream>>>(
            x, (const _Float16*)nullptr, wpack, scale, zp, bias, out);
    }
}
